// Classifier_3100966387978
// MI455X (gfx1250) — matrix operand order repaired and checked by reading, not compiled, not run on hardware
//
#include <hip/hip_runtime.h>
#include <math.h>

// ---------------- problem constants ----------------
#define NTOK   65536
#define DDIM   512
#define EEXP   16
#define HHID   128
#define KTOP   12
#define ODIM   8
#define TOKB   32            // tokens per MoE block (2 M-tiles)

// ---------------- vector types ----------------
typedef __attribute__((ext_vector_type(16))) __bf16 bf16x16;
typedef __attribute__((ext_vector_type(8)))  __bf16 bf16x8;
typedef __attribute__((ext_vector_type(4)))  __bf16 bf16x4;
typedef __attribute__((ext_vector_type(8)))  float  f32x8;
typedef __attribute__((ext_vector_type(4)))  float  f32x4;

// LDS row strides (elements), padded +8 bf16 (=16B) to break 64-bank alignment
#define XS_LD 520      // 32 rows x 512 (+8 pad) bf16
#define HW_LD 2056     // 32 rows x 2048 (+8 pad) bf16
#define SMEM_BYTES (TOKB*XS_LD*2 + TOKB*HW_LD*2 + TOKB*16*4 + TOKB*512*4)

// Build an A-fragment (16x32 bf16, row M = lane&15) from a bf16 row in LDS.
// Layout: lanes 0-15 elems[0..15]=K(kb..kb+15);
// lanes 16-31 elems[0..15]=K(kb+16..kb+31).
__device__ __forceinline__ bf16x16 lds_a_frag(const __bf16* row, int kbase, int s) {
    bf16x8 lo = *(const bf16x8*)(row + kbase + s * 16);
    bf16x8 hi = *(const bf16x8*)(row + kbase + s * 16 + 8);
    bf16x16 r;
#pragma unroll
    for (int i = 0; i < 8; ++i) { r[i] = lo[i]; r[8 + i] = hi[i]; }
    return r;
}

// ---------------- pre-pass: f32 -> bf16 fragment-major weights ----------------
// dst flat index = mat*fc + ((jn*nChunks + c)*32 + lane)*16 + t
// element = src[mat][k = c*32 + (lane>>4)*16 + t][n = jn*16 + (lane&15)]
__global__ void frag_convert(const float* __restrict__ src, __bf16* __restrict__ dst,
                             int Kdim, int NdimPad, int NdimValid, int srcStride,
                             int nMat, long long matStride,
                             const int* __restrict__ modSel, long long modStride) {
    long long idx = (long long)blockIdx.x * blockDim.x + threadIdx.x;
    int nChunks = Kdim >> 5;
    long long fc = (long long)(NdimPad >> 4) * nChunks * 512;
    if (idx >= (long long)nMat * fc) return;
    if (modSel) src += (long long)modSel[0] * modStride;
    int mat = (int)(idx / fc);
    long long rem = idx - (long long)mat * fc;
    int t = (int)(rem & 15);
    int lane = (int)((rem >> 4) & 31);
    long long rest = rem >> 9;
    int c  = (int)(rest % nChunks);
    int jn = (int)(rest / nChunks);
    int k = c * 32 + (lane >> 4) * 16 + t;
    int n = jn * 16 + (lane & 15);
    float v = (n < NdimValid) ? src[(long long)mat * matStride + (long long)k * srcStride + n] : 0.0f;
    dst[idx] = (__bf16)v;
}

__global__ void zero16(float* p) { if (threadIdx.x < 16) p[threadIdx.x] = 0.0f; }

// ---------------- gating: logits WMMA + top-12 softmax ----------------
__global__ __launch_bounds__(256) void gate_kernel(const float* __restrict__ x,
                                                   const __bf16* __restrict__ Wgf,
                                                   float* __restrict__ gates,
                                                   float* __restrict__ load) {
    __shared__ float lt[8 * 16 * 16];
    __shared__ float cnt[16];
    int tid = threadIdx.x, lane = tid & 31, wave = tid >> 5;
    if (tid < 16) cnt[tid] = 0.0f;

    int tokBase = blockIdx.x * 128 + wave * 16;
    const float* xr = x + (size_t)(tokBase + (lane & 15)) * DDIM;
    int s = lane >> 4;
    f32x8 acc = {};
#pragma unroll
    for (int c16 = 0; c16 < 16; ++c16) {
        const float* p = xr + c16 * 32 + s * 16;
        f32x4 f0 = *(const f32x4*)(p);
        f32x4 f1 = *(const f32x4*)(p + 4);
        f32x4 f2 = *(const f32x4*)(p + 8);
        f32x4 f3 = *(const f32x4*)(p + 12);
        bf16x16 a;
#pragma unroll
        for (int i = 0; i < 4; ++i) {
            a[i]      = (__bf16)f0[i];
            a[4 + i]  = (__bf16)f1[i];
            a[8 + i]  = (__bf16)f2[i];
            a[12 + i] = (__bf16)f3[i];
        }
        bf16x16 b = *(const bf16x16*)(Wgf + c16 * 512 + lane * 16);
        acc = __builtin_amdgcn_wmma_f32_16x16x32_bf16(false, a, false, b,
                                                      (short)0, acc, false, false);
    }
#pragma unroll
    for (int r = 0; r < 8; ++r)
        lt[wave * 256 + (r + 8 * s) * 16 + (lane & 15)] = acc[r];
    __syncthreads();

    if (tid < 128) {
        const float* row = lt + (tid >> 4) * 256 + (tid & 15) * 16;
        float v[16];
#pragma unroll
        for (int e = 0; e < 16; ++e) v[e] = row[e];
        unsigned excl = 0;
#pragma unroll
        for (int it = 0; it < EEXP - KTOP; ++it) {         // drop 4 smallest
            float mn = 3.4e38f; int mi = 0;
#pragma unroll
            for (int e = 0; e < 16; ++e)
                if (!((excl >> e) & 1u) && v[e] < mn) { mn = v[e]; mi = e; }
            excl |= 1u << mi;
        }
        float mx = -3.4e38f;
#pragma unroll
        for (int e = 0; e < 16; ++e)
            if (!((excl >> e) & 1u)) mx = fmaxf(mx, v[e]);
        float w[16]; float sum = 0.0f;
#pragma unroll
        for (int e = 0; e < 16; ++e) {
            float t = ((excl >> e) & 1u) ? 0.0f : __expf(v[e] - mx);
            w[e] = t; sum += t;
        }
        float inv = 1.0f / sum;
        float* g = gates + (size_t)(blockIdx.x * 128 + tid) * 16;
#pragma unroll
        for (int e = 0; e < 16; ++e) g[e] = w[e] * inv;
#pragma unroll
        for (int e = 0; e < 16; ++e)
            if (!((excl >> e) & 1u)) atomicAdd(&cnt[e], 1.0f);
    }
    __syncthreads();
    if (tid < 16) atomicAdd(&load[tid], cnt[tid]);
}

// ---------------- main MoE + head kernel: 32 tokens / block, 8 waves ----------------
__global__ __launch_bounds__(256) void moe_kernel(const float* __restrict__ x,
                                                  const float* __restrict__ gatesIn,
                                                  const __bf16* __restrict__ W1f,
                                                  const __bf16* __restrict__ W2f,
                                                  const __bf16* __restrict__ Woutf,
                                                  const float* __restrict__ b1g,
                                                  const float* __restrict__ b2g,
                                                  const float* __restrict__ boutg,
                                                  float* __restrict__ y) {
    extern __shared__ __align__(16) char smem[];
    __bf16* xs = (__bf16*)smem;                       // [32][XS_LD] x strip -> later h2 (bf16)
    __bf16* hw = xs + TOKB * XS_LD;                   // [32][HW_LD] gated hidden, K = e*128+h
    float*  gt = (float*)(hw + TOKB * HW_LD);         // [32][16] gates tile
    float*  bd = gt + TOKB * 16;                      // [32][512] gates @ b2

    int tid = threadIdx.x, lane = tid & 31, wave = tid >> 5;
    int s = lane >> 4;
    int tokBase = blockIdx.x * TOKB;

    // gates tile: async global->LDS copy (ASYNCcnt path), 32x16 f32 = 2 KB
    if (wave == 0) {
        unsigned ldsBase = (unsigned)(size_t)(void*)gt;
        const void* gsrc = (const void*)(gatesIn + (size_t)tokBase * 16);
#pragma unroll
        for (int i = 0; i < 4; ++i) {
            unsigned voff = (unsigned)(lane * 16 + i * 512);
            unsigned ldst = ldsBase + voff;
            asm volatile("global_load_async_to_lds_b128 %0, %1, %2"
                         :: "v"(ldst), "v"(voff), "s"(gsrc) : "memory");
        }
        asm volatile("s_wait_asynccnt 0x0" ::: "memory");
    }

    // stage x strip (f32 -> bf16): 32x512 = 4096 float4
    for (int i = tid; i < 4096; i += 256) {
        int row = i >> 7;
        int col = (i & 127) * 4;
        f32x4 v = *(const f32x4*)(x + (size_t)(tokBase + row) * DDIM + col);
        bf16x4 b;
#pragma unroll
        for (int j = 0; j < 4; ++j) b[j] = (__bf16)v[j];
        *(bf16x4*)(xs + row * XS_LD + col) = b;
    }
    __syncthreads();

    // bd[tok][d] = sum_e gt[tok][e] * b2[e][d]
    {
        int tok = tid >> 3;
        int d0 = (tid & 7) * 64;
        for (int dd = 0; dd < 64; ++dd) {
            int d = d0 + dd;
            float a = 0.0f;
#pragma unroll
            for (int e = 0; e < 16; ++e) a += gt[tok * 16 + e] * b2g[e * DDIM + d];
            bd[tok * 512 + d] = a;
        }
    }
    __syncthreads();

    const __bf16* row0 = xs + (lane & 15) * XS_LD;            // M-tile 0
    const __bf16* row1 = xs + (16 + (lane & 15)) * XS_LD;     // M-tile 1

    // GEMM1: per (expert e, n-tile jn), 2 M-tiles per B fragment
    for (int t = wave; t < 128; t += 8) {
        int e = t >> 3, jn = t & 7;
        f32x8 c0 = {}, c1 = {};
        const __bf16* Bb = W1f + ((size_t)((e * 8 + jn) * 16)) * 512 + lane * 16;
#pragma unroll
        for (int c16 = 0; c16 < 16; ++c16) {
            bf16x16 b = *(const bf16x16*)(Bb + (size_t)c16 * 512);
            bf16x16 a0 = lds_a_frag(row0, c16 * 32, s);
            c0 = __builtin_amdgcn_wmma_f32_16x16x32_bf16(false, a0, false, b,
                                                         (short)0, c0, false, false);
            bf16x16 a1 = lds_a_frag(row1, c16 * 32, s);
            c1 = __builtin_amdgcn_wmma_f32_16x16x32_bf16(false, a1, false, b,
                                                         (short)0, c1, false, false);
        }
        int col = jn * 16 + (lane & 15);
        float b1v = b1g[e * HHID + col];
#pragma unroll
        for (int r = 0; r < 8; ++r) {
            int tok = r + 8 * s;
            float v = c0[r] + b1v;
            v = v > 0.0f ? v : 0.0f;
            v *= gt[tok * 16 + e];
            hw[tok * HW_LD + e * HHID + col] = (__bf16)v;
            int tok1 = tok + 16;
            float v1 = c1[r] + b1v;
            v1 = v1 > 0.0f ? v1 : 0.0f;
            v1 *= gt[tok1 * 16 + e];
            hw[tok1 * HW_LD + e * HHID + col] = (__bf16)v1;
        }
    }
    __syncthreads();

    // GEMM2: out tiles = hw[32x2048] @ W2f[2048x512]; epilogue -> h2 bf16 into xs
    const __bf16* hrow0 = hw + (lane & 15) * HW_LD;
    const __bf16* hrow1 = hw + (16 + (lane & 15)) * HW_LD;
    for (int jn2 = wave; jn2 < 32; jn2 += 8) {
        f32x8 c0 = {}, c1 = {};
        const __bf16* Bb = W2f + ((size_t)(jn2 * 64)) * 512 + lane * 16;
#pragma unroll 4
        for (int c2 = 0; c2 < 64; ++c2) {
            bf16x16 b = *(const bf16x16*)(Bb + (size_t)c2 * 512);
            bf16x16 a0 = lds_a_frag(hrow0, c2 * 32, s);
            c0 = __builtin_amdgcn_wmma_f32_16x16x32_bf16(false, a0, false, b,
                                                         (short)0, c0, false, false);
            bf16x16 a1 = lds_a_frag(hrow1, c2 * 32, s);
            c1 = __builtin_amdgcn_wmma_f32_16x16x32_bf16(false, a1, false, b,
                                                         (short)0, c1, false, false);
        }
        int col = jn2 * 16 + (lane & 15);
#pragma unroll
        for (int r = 0; r < 8; ++r) {
            int tok = r + 8 * s;
            float o = c0[r] + bd[tok * 512 + col];
            o = o > 0.0f ? o : 0.0f;
            o += (float)xs[tok * XS_LD + col];            // residual
            xs[tok * XS_LD + col] = (__bf16)o;            // h2 (overwrites x)
            int tok1 = tok + 16;
            float o1 = c1[r] + bd[tok1 * 512 + col];
            o1 = o1 > 0.0f ? o1 : 0.0f;
            o1 += (float)xs[tok1 * XS_LD + col];
            xs[tok1 * XS_LD + col] = (__bf16)o1;
        }
    }
    __syncthreads();

    // GEMM3 (waves 0,1): y = h2 @ Wout(+pad) + bout, one M-tile per wave
    if (wave < 2) {
        const __bf16* hr = xs + (16 * wave + (lane & 15)) * XS_LD;
        f32x8 c = {};
#pragma unroll
        for (int c16 = 0; c16 < 16; ++c16) {
            bf16x16 a = lds_a_frag(hr, c16 * 32, s);
            bf16x16 b = *(const bf16x16*)(Woutf + (size_t)c16 * 512 + lane * 16);
            c = __builtin_amdgcn_wmma_f32_16x16x32_bf16(false, a, false, b,
                                                        (short)0, c, false, false);
        }
        int col = lane & 15;
        if (col < ODIM) {
            float bo = boutg[col];
#pragma unroll
            for (int r = 0; r < 8; ++r)
                y[(size_t)(tokBase + 16 * wave + r + 8 * s) * ODIM + col] = c[r] + bo;
        }
    }
}

// ---------------- host launch ----------------
extern "C" void kernel_launch(void* const* d_in, const int* in_sizes, int n_in,
                              void* d_out, int out_size, void* d_ws, size_t ws_size,
                              hipStream_t stream) {
    const float* x    = (const float*)d_in[0];
    const int*   mod  = (const int*)d_in[1];
    const float* Wg   = (const float*)d_in[2];
    const float* W1   = (const float*)d_in[3];
    const float* b1   = (const float*)d_in[4];
    const float* W2   = (const float*)d_in[5];
    const float* b2   = (const float*)d_in[6];
    const float* Wout = (const float*)d_in[7];
    const float* bout = (const float*)d_in[8];

    float* y     = (float*)d_out;                       // [N,8]
    float* gates = y + (size_t)NTOK * ODIM;             // [N,16]
    float* load  = gates + (size_t)NTOK * EEXP;         // [16]

    __bf16* W1f   = (__bf16*)d_ws;                      // 1,048,576 elems
    __bf16* W2f   = W1f + (size_t)1048576;              // 1,048,576 elems
    __bf16* Wgf   = W2f + (size_t)1048576;              // 8,192 elems
    __bf16* Woutf = Wgf + (size_t)8192;                 // 8,192 elems

    // weight fragment conversions
    frag_convert<<<4096, 256, 0, stream>>>(W1, W1f, 512, 128, 128, 128,
                                           16, (long long)512 * 128, nullptr, 0);
    frag_convert<<<4096, 256, 0, stream>>>(W2, W2f, 2048, 512, 512, 512,
                                           1, 0, nullptr, 0);
    frag_convert<<<32, 256, 0, stream>>>(Wg, Wgf, 512, 16, 16, 16,
                                         1, 0, mod, (long long)DDIM * EEXP);
    frag_convert<<<32, 256, 0, stream>>>(Wout, Woutf, 512, 16, 8, 8,
                                         1, 0, nullptr, 0);
    zero16<<<1, 32, 0, stream>>>(load);

    gate_kernel<<<NTOK / 128, 256, 0, stream>>>(x, Wgf, gates, load);

    moe_kernel<<<NTOK / TOKB, 256, SMEM_BYTES, stream>>>(x, gates, W1f, W2f, Woutf,
                                                         b1, b2, bout, y);
}
